// LinkPredictorMLP_31344671326723
// MI455X (gfx1250) — matrix operand order repaired and checked by reading, not compiled, not run on hardware
//
#include <hip/hip_runtime.h>

typedef __attribute__((ext_vector_type(16))) _Float16     v16h;
typedef __attribute__((ext_vector_type(8)))  float        v8f;
typedef __attribute__((ext_vector_type(4)))  unsigned int v4u;
typedef __attribute__((ext_vector_type(8)))  int          v8i;
typedef __attribute__((ext_vector_type(4)))  int          v4i;

#define DIM        128          // embedding dim
#define TWOD       256          // concat dim / W1 rows (K)
#define HID        256          // hidden dim (N)
#define NT         16           // N tiles of 16
#define KT         8            // K tiles of 32
#define MSET       2            // A-tile sets per wave (32 edges/wave/iter)
#define WAVES      8            // waves per block
#define ITERS      4            // 32-edge tiles per wave (amortize W1 staging)
#define THREADS    256
#define T_SMALL    0.05f

// LDS: W1 as f16, pre-swizzled into per-lane WMMA-B layout:
// index = ((nt*KT + kt)*32 + lane)*16 + j  ;  16 halves per lane = one v16h
#define LDS_B_HALFS (NT * KT * 32 * 16)     // 65536 halves = 128 KB
#define W1_IMG_BYTES (LDS_B_HALFS * 2)      // 128 KB swizzled image
#define LDS_BYTES   (W1_IMG_BYTES + HID * 4 + HID * 4)

// ---------------- prep: build swizzled f16 W1 image in workspace ----------
__global__ __launch_bounds__(THREADS)
void prep_w1(const float* __restrict__ W1, _Float16* __restrict__ img)
{
    const int g  = blockIdx.x * THREADS + threadIdx.x;   // one float4-quad each
    const int k  = g >> 6;                               // 64 quads per k-row
    const int n0 = (g & 63) * 4;
    const float4 f = *(const float4*)(W1 + k * HID + n0);
    const int nt   = n0 >> 4;
    const int kt   = k >> 5;
    const int lnhi = (k >> 4) & 1;
    const int j    = k & 15;
    const int base = (((nt * KT + kt) * 32) + lnhi * 16 + (n0 & 15)) * 16 + j;
    img[base]      = (_Float16)f.x;
    img[base + 16] = (_Float16)f.y;
    img[base + 32] = (_Float16)f.z;
    img[base + 48] = (_Float16)f.w;
}

// ---------------- main: WMMA link-predictor MLP ---------------------------
__global__ __launch_bounds__(THREADS)
void link_mlp_wmma(const float* __restrict__ emb,
                   const int*   __restrict__ eidx,
                   const float* __restrict__ W1,
                   const float* __restrict__ b1,
                   const float* __restrict__ W2,
                   const float* __restrict__ b2,
                   const _Float16* __restrict__ w1img,   // swizzled image (d_ws)
                   float*       __restrict__ out,
                   int E, int useWs)
{
    extern __shared__ char smem[];
    _Float16* bsm = (_Float16*)smem;
    float*    b1s = (float*)(smem + W1_IMG_BYTES);
    float*    w2s = b1s + HID;

    const int tid = threadIdx.x;

    if (useWs) {
        // ---- TDM: DMA the prebuilt 128KB swizzled W1 image into LDS ----
        // 2-D descriptor per ISA 8.3/8.4: one row of 32768 x 4B elements.
        if (tid < 32) {                       // wave 0 only (EXEC ignored)
            const unsigned long long ga = (unsigned long long)(uintptr_t)w1img;
            const unsigned int ldsa     = (unsigned int)(uintptr_t)smem; // LDS byte offset
            v4u g0;
            g0[0] = 1u;                                  // count=1 (valid D#)
            g0[1] = ldsa;                                // lds_addr
            g0[2] = (unsigned int)ga;                    // global_addr[31:0]
            g0[3] = (unsigned int)((ga >> 32) & 0x1FFFFFFu) | 0x80000000u; // addr[56:32] | type=2
            v8i g1;
            g1[0] = (int)(2u << 16);                     // data_size=4B; no mask/pad/iter
            g1[1] = (int)0x80000000u;                    // tensor_dim0 lo16=0x8000 (32768)
            g1[2] = (int)0x00010000u;                    // tensor_dim0 hi=0; tensor_dim1=1
            g1[3] = (int)0x80000000u;                    // tile_dim0 = 32768
            g1[4] = 1;                                   // tile_dim1 = 1; tile_dim2 = 0
            g1[5] = 32768;                               // tensor_dim0_stride lo
            g1[6] = 0;                                   // stride hi / dim1_stride lo
            g1[7] = 0;
            v4i z4; z4[0] = 0; z4[1] = 0; z4[2] = 0; z4[3] = 0;
            v8i z8; z8[0] = 0; z8[1] = 0; z8[2] = 0; z8[3] = 0;
                    z8[4] = 0; z8[5] = 0; z8[6] = 0; z8[7] = 0;
            __builtin_amdgcn_tensor_load_to_lds(g0, g1, z4, z4, z8, 0);
            __builtin_amdgcn_s_wait_tensorcnt(0);
        }
    } else {
        // ---- Fallback: stage W1 (f32 -> f16) into WMMA-B LDS layout ----
        for (int g = tid; g < (TWOD * HID / 4); g += THREADS) {
            const int k  = g >> 6;
            const int n0 = (g & 63) * 4;
            const float4 f = *(const float4*)(W1 + k * HID + n0);
            const int nt   = n0 >> 4;
            const int kt   = k >> 5;
            const int lnhi = (k >> 4) & 1;
            const int j    = k & 15;
            const int base = (((nt * KT + kt) * 32) + lnhi * 16 + (n0 & 15)) * 16 + j;
            bsm[base]      = (_Float16)f.x;
            bsm[base + 16] = (_Float16)f.y;
            bsm[base + 32] = (_Float16)f.z;
            bsm[base + 48] = (_Float16)f.w;
        }
    }
    if (tid < HID) { b1s[tid] = b1[tid]; w2s[tid] = W2[tid]; }
    __syncthreads();

    const int lane     = tid & 31;
    const int wv       = tid >> 5;
    const int tileBase = blockIdx.x * (WAVES * ITERS);   // 32-edge tiles
    const int khalf    = (lane >> 4) * 16;
    const float b2v    = b2[0];

    for (int it = 0; it < ITERS; ++it) {
        const int etile = tileBase + it * WAVES + wv;
        if (etile * 32 >= E) break;        // tail (no barriers below)

        // ---- Gather A: 2 sets x 16 edges x 256 features, f32 -> f16 ----
        // A (16x32 f16): lanes 0-15 rows M=0..15 hold K = khalf+0..15
        // (halves 0-15), khalf = 16*(lane>=16), matching B's contiguous order.
        v16h areg[MSET][KT];
#pragma unroll
        for (int s = 0; s < MSET; ++s) {
            const int e    = etile * 32 + s * 16 + (lane & 15);
            const int ridx = eidx[e];
            const int cidx = eidx[E + e];
#pragma unroll
            for (int kt = 0; kt < KT; ++kt) {
                const int    nid = (kt < 4) ? ridx : cidx;  // K<128: row else col
                const float* bse = emb + (long)nid * DIM + (kt & 3) * 32 + khalf;
                const float4* p  = (const float4*)bse;      // 32B-aligned
                const float4 f0 = p[0], f1 = p[1];          // K = khalf   + 0..7
                const float4 f2 = p[2], f3 = p[3];          // K = khalf+8 + 0..7
                v16h a;
                a[0]=(_Float16)f0.x;  a[1]=(_Float16)f0.y;  a[2]=(_Float16)f0.z;  a[3]=(_Float16)f0.w;
                a[4]=(_Float16)f1.x;  a[5]=(_Float16)f1.y;  a[6]=(_Float16)f1.z;  a[7]=(_Float16)f1.w;
                a[8]=(_Float16)f2.x;  a[9]=(_Float16)f2.y;  a[10]=(_Float16)f2.z; a[11]=(_Float16)f2.w;
                a[12]=(_Float16)f3.x; a[13]=(_Float16)f3.y; a[14]=(_Float16)f3.z; a[15]=(_Float16)f3.w;
                areg[s][kt] = a;
            }
        }

        // ---- GEMM: 16 n-tiles x 8 k-tiles; each B ds-load feeds TWO wmma.
        //      Fused epilogue: +b1, relu, * W2 accumulate. ----
        float acc[MSET][8];
#pragma unroll
        for (int s = 0; s < MSET; ++s)
#pragma unroll
            for (int v = 0; v < 8; ++v) acc[s][v] = 0.f;

        for (int nt = 0; nt < NT; ++nt) {
            v8f c0 = {}, c1 = {};
            const _Float16* bp = bsm + ((nt * KT) * 32 + lane) * 16;
#pragma unroll
            for (int kt = 0; kt < KT; ++kt) {
                const v16h bmat = *(const v16h*)(bp + kt * (32 * 16));
                c0 = __builtin_amdgcn_wmma_f32_16x16x32_f16(
                         false, areg[0][kt], false, bmat, (short)0, c0, false, false);
                c1 = __builtin_amdgcn_wmma_f32_16x16x32_f16(
                         false, areg[1][kt], false, bmat, (short)0, c1, false, false);
            }
            const int   n   = nt * 16 + (lane & 15);
            const float b1v = b1s[n];
            const float w2v = w2s[n];
#pragma unroll
            for (int v = 0; v < 8; ++v) {        // C layout: VGPR v -> row
                float t0 = c0[v] + b1v;          //  M = v + 8*(lane>=16)
                float t1 = c1[v] + b1v;
                t0 = t0 > 0.f ? t0 : 0.f;        // relu(hidden)
                t1 = t1 > 0.f ? t1 : 0.f;
                acc[0][v] += t0 * w2v;           // second-layer dot, partial
                acc[1][v] += t1 * w2v;
            }
        }

        // ---- Reduce over the 16 lanes of each half (over N), finish ----
#pragma unroll
        for (int s = 0; s < MSET; ++s) {
#pragma unroll
            for (int v = 0; v < 8; ++v) {
                float r = acc[s][v];
                for (int m = 1; m < 16; m <<= 1) r += __shfl_xor(r, m, 32);
                acc[s][v] = r;
            }
            if ((lane & 15) == 0) {
                const int mbase = (lane >> 4) * 8; // lane0: rows 0..7, lane16: 8..15
#pragma unroll
                for (int v = 0; v < 8; ++v) {
                    float raw = acc[s][v] + b2v;
                    float w   = raw > 0.f ? raw : 0.f;
                    out[etile * 32 + s * 16 + mbase + v] = (w < T_SMALL) ? 0.f : w;
                }
            }
        }
    }
}

extern "C" void kernel_launch(void* const* d_in, const int* in_sizes, int n_in,
                              void* d_out, int out_size, void* d_ws, size_t ws_size,
                              hipStream_t stream) {
    const float* emb = (const float*)d_in[0];
    const int*   idx = (const int*)  d_in[1];
    const float* W1  = (const float*)d_in[2];
    const float* b1  = (const float*)d_in[3];
    const float* W2  = (const float*)d_in[4];
    const float* b2  = (const float*)d_in[5];
    float*       out = (float*)d_out;

    const int E      = in_sizes[1] / 2;                  // edge_index is (2,E)
    const int etiles = (E + 31) / 32;                    // 32 edges per wave-iter
    const int blocks = (etiles + WAVES * ITERS - 1) / (WAVES * ITERS);

    const int useWs  = (ws_size >= (size_t)W1_IMG_BYTES) ? 1 : 0;
    if (useWs) {
        // Build the swizzled f16 W1 image once; every block then DMAs it
        // into LDS via the Tensor Data Mover instead of re-swizzling.
        prep_w1<<<(TWOD * HID / 4) / THREADS, THREADS, 0, stream>>>(
            W1, (_Float16*)d_ws);
    }

    // Opt in to >64KB dynamic LDS (WGP supports 320KB); capture-safe (no
    // stream/alloc/sync APIs).
    (void)hipFuncSetAttribute((const void*)link_mlp_wmma,
                              hipFuncAttributeMaxDynamicSharedMemorySize,
                              LDS_BYTES);

    link_mlp_wmma<<<blocks, THREADS, LDS_BYTES, stream>>>(
        emb, idx, W1, b1, W2, b2, (const _Float16*)d_ws, out, E, useWs);
}
